// protonet_Model_10462540333796
// MI455X (gfx1250) — matrix operand order repaired and checked by reading, not compiled, not run on hardware
//
#include <hip/hip_runtime.h>
#include <hip/hip_bf16.h>

typedef __attribute__((ext_vector_type(16))) _Float16 v16h;
typedef __attribute__((ext_vector_type(8)))  _Float16 v8h;
typedef __attribute__((ext_vector_type(8)))  float    v8f;
typedef __attribute__((ext_vector_type(4)))  float    v4f;
typedef __attribute__((vector_size(16)))     int      v4i_;

#define INPUT_DIM 128
#define AE_HID    64
#define HIDDEN    25
#define H1D       40
#define H2D       25
#define OUTD      10
#define NCLASS    2
#define BN_EPS    1e-5f
#define COS_EPS   1e-8f

#define TILE_M 128
#define XSTR   136   // halfs per row for X / C2 tile (128 + 8 pad)
#define HSTR   72    // halfs per row for H / W2t    (64 + 8 pad)

// CDNA5 async global->LDS path (guarded; sync fallback keeps compile safe)
#if defined(__gfx1250__) && __has_builtin(__builtin_amdgcn_global_load_async_to_lds_b128)
#define HAVE_ASYNC_LDS 1
typedef __attribute__((address_space(1))) v4i_ g_v4i;   // global 16B vector
typedef __attribute__((address_space(3))) v4i_ l_v4i;   // LDS    16B vector
#else
#define HAVE_ASYNC_LDS 0
#endif

// float offsets inside the small-weight LDS region
enum {
  O_WIN  = 0,                // 128*25
  O_BIN  = O_WIN + 3200,     // 25
  O_GIN  = O_BIN + 25,  O_BEIN = O_GIN + 25,  O_MIN = O_BEIN + 25,  O_VIN = O_MIN + 25,
  O_WH1  = O_VIN + 25,       // 25*40
  O_BH1  = O_WH1 + 1000, O_GH1 = O_BH1 + 40,  O_BEH1 = O_GH1 + 40, O_MH1 = O_BEH1 + 40, O_VH1 = O_MH1 + 40,
  O_WH2  = O_VH1 + 40,       // 40*25
  O_BH2  = O_WH2 + 1000, O_GH2 = O_BH2 + 25,  O_BEH2 = O_GH2 + 25, O_MH2 = O_BEH2 + 25, O_VH2 = O_MH2 + 25,
  O_WOUT = O_VH2 + 25,       // 25*10
  O_BOUT = O_WOUT + 250,     // 10
  O_AEB1 = O_BOUT + 10,      // 64
  O_AEB2 = O_AEB1 + 64,      // 128
  O_PN   = O_AEB2 + 128,     // 2*10
  SF_TOTAL = O_PN + 20
};

#define XF_BYTES  (TILE_M*INPUT_DIM*4)   // f32 async staging region (65536 B)
#define LDS_HALFS (TILE_M*XSTR + 64*XSTR + TILE_M*HSTR + 128*HSTR)
#define LDS_BYTES (XF_BYTES + LDS_HALFS*2 + SF_TOTAL*4)

struct Params {
  const float *ae_w1, *ae_b1, *ae_w2, *ae_b2;
  const float *w_in,  *b_in,  *g_in,  *be_in, *m_in, *v_in;
  const float *w_h1,  *b_h1,  *g_h1,  *be_h1, *m_h1, *v_h1;
  const float *w_h2,  *b_h2,  *g_h2,  *be_h2, *m_h2, *v_h2;
  const float *w_out, *b_out;
};

__device__ __forceinline__ v16h cat16(v8h lo, v8h hi) {
  return __builtin_shufflevector(lo, hi, 0,1,2,3,4,5,6,7,8,9,10,11,12,13,14,15);
}

__device__ __forceinline__ void wait_asynccnt0() {
#if HAVE_ASYNC_LDS
#if __has_builtin(__builtin_amdgcn_s_wait_asynccnt)
  __builtin_amdgcn_s_wait_asynccnt(0);
#else
  asm volatile("s_wait_asynccnt 0" ::: "memory");
#endif
#endif
}

__global__ __launch_bounds__(256)
void embed_kernel(const float* __restrict__ X, int nrows, Params p,
                  float* __restrict__ emb_out,       // mode 0: [nrows][10]
                  const float* __restrict__ pnG,     // mode 1: normalized protos [2][10]
                  float* __restrict__ cos_out,       // mode 1: [nrows][2]
                  int mode)
{
  extern __shared__ char smem_raw[];
  float*    sXF = (float*)smem_raw;                   // [128][128] f32 async staging
  _Float16* sX  = (_Float16*)(smem_raw + XF_BYTES);   // [128][XSTR]; becomes C2 after layer2
  _Float16* sW1 = sX  + TILE_M*XSTR;                  // [64][XSTR]   W1^T (n-major, K=128)
  _Float16* sH  = sW1 + 64*XSTR;                      // [128][HSTR]  layer1 output (relu)
  _Float16* sW2 = sH  + TILE_M*HSTR;                  // [128][HSTR]  W2^T (n-major, K=64)
  float*    sF  = (float*)(sW2 + 128*HSTR);           // small weights

  const int tid  = threadIdx.x;
  const long long row0 = (long long)blockIdx.x * TILE_M;

  // speculative prefetch of next tile (global_prefetch_b8)
  {
    long long nb = row0 + TILE_M;
    if (nb < nrows) __builtin_prefetch(X + nb*INPUT_DIM + tid*64, 0, 0);
  }

#if HAVE_ASYNC_LDS
  // ---- fire the whole X tile as async global->LDS (raw f32), 16 B/lane/iter ----
  {
    #pragma unroll
    for (int j = 0; j < 16; ++j) {
      int e4 = j*256 + tid;              // 16-byte chunk id within tile
      int r  = e4 >> 5;                  // row (32 chunks of 4 floats per row)
      long long gr = row0 + r;
      if (gr < nrows) {
        const float* g = X + gr*INPUT_DIM + (e4 & 31)*4;
        __builtin_amdgcn_global_load_async_to_lds_b128(
            (g_v4i*)g, (l_v4i*)(sXF + e4*4), 0, 0);
      }
    }
  }
#endif

  // ---- stage weights (transposed, f16) while the tile is in flight ----
  for (int i = tid; i < 64*128; i += 256) {        // sW1[n][k] = ae_w1[k][n]
    int n = i >> 7, k = i & 127;
    sW1[n*XSTR + k] = (_Float16)p.ae_w1[k*AE_HID + n];
  }
  for (int i = tid; i < 128*64; i += 256) {        // sW2[n][k] = ae_w2[k][n]
    int n = i >> 6, k = i & 63;
    sW2[n*HSTR + k] = (_Float16)p.ae_w2[k*INPUT_DIM + n];
  }
  // ---- stage small weights ----
  {
    auto cp = [&](int off, const float* src, int n) {
      for (int i = tid; i < n; i += 256) sF[off + i] = src[i];
    };
    cp(O_WIN, p.w_in, 3200);  cp(O_BIN, p.b_in, 25);
    cp(O_GIN, p.g_in, 25);    cp(O_BEIN, p.be_in, 25); cp(O_MIN, p.m_in, 25); cp(O_VIN, p.v_in, 25);
    cp(O_WH1, p.w_h1, 1000);  cp(O_BH1, p.b_h1, 40);
    cp(O_GH1, p.g_h1, 40);    cp(O_BEH1, p.be_h1, 40); cp(O_MH1, p.m_h1, 40); cp(O_VH1, p.v_h1, 40);
    cp(O_WH2, p.w_h2, 1000);  cp(O_BH2, p.b_h2, 25);
    cp(O_GH2, p.g_h2, 25);    cp(O_BEH2, p.be_h2, 25); cp(O_MH2, p.m_h2, 25); cp(O_VH2, p.v_h2, 25);
    cp(O_WOUT, p.w_out, 250); cp(O_BOUT, p.b_out, 10);
    cp(O_AEB1, p.ae_b1, 64);  cp(O_AEB2, p.ae_b2, 128);
    if (pnG) cp(O_PN, pnG, NCLASS*OUTD);
  }

#if HAVE_ASYNC_LDS
  // ---- drain ASYNCcnt, then convert own chunks f32 -> f16 (lane-local, no barrier) ----
  wait_asynccnt0();
  #pragma unroll
  for (int j = 0; j < 16; ++j) {
    int e4 = j*256 + tid;
    int r  = e4 >> 5, c = (e4 & 31)*4;
    long long gr = row0 + r;
    v4f v = {0.f, 0.f, 0.f, 0.f};
    if (gr < nrows) v = *(const v4f*)(sXF + e4*4);
    sX[r*XSTR + c + 0] = (_Float16)v[0];
    sX[r*XSTR + c + 1] = (_Float16)v[1];
    sX[r*XSTR + c + 2] = (_Float16)v[2];
    sX[r*XSTR + c + 3] = (_Float16)v[3];
  }
#else
  // ---- synchronous fallback: global f32 -> f16 LDS ----
  for (int i = tid; i < 128*32; i += 256) {
    int r = i >> 5, c = (i & 31) * 4;
    long long gr = row0 + r;
    v4f v = {0.f, 0.f, 0.f, 0.f};
    if (gr < nrows) v = *(const v4f*)(X + gr*INPUT_DIM + c);
    sX[r*XSTR + c + 0] = (_Float16)v[0];
    sX[r*XSTR + c + 1] = (_Float16)v[1];
    sX[r*XSTR + c + 2] = (_Float16)v[2];
    sX[r*XSTR + c + 3] = (_Float16)v[3];
  }
#endif
  __syncthreads();

  const int  wave  = tid >> 5;
  const int  lane  = tid & 31;
  const int  lhalf = lane & 15;
  const bool hiH   = (lane >= 16);

  // ======== Layer 1: H[128,64] = relu(X[128,128] @ W1 + b1), WMMA f16 ========
  {
    v8f acc[4];
    #pragma unroll
    for (int n = 0; n < 4; ++n) {
      float bv = sF[O_AEB1 + n*16 + lhalf];
      v8f t = {bv,bv,bv,bv,bv,bv,bv,bv};
      acc[n] = t;
    }
    #pragma unroll
    for (int k0 = 0; k0 < INPUT_DIM; k0 += 32) {
      const _Float16* pa = sX + (wave*16 + lhalf)*XSTR + k0 + (hiH ? 8 : 0);
      v16h a = cat16(*(const v8h*)pa, *(const v8h*)(pa + 16));
      #pragma unroll
      for (int n = 0; n < 4; ++n) {
        const _Float16* pb = sW1 + (n*16 + lhalf)*XSTR + k0 + (hiH ? 8 : 0);
        v16h b = cat16(*(const v8h*)pb, *(const v8h*)(pb + 16));
        acc[n] = __builtin_amdgcn_wmma_f32_16x16x32_f16(false, a, false, b,
                                                        (short)0, acc[n], false, false);
      }
    }
    #pragma unroll
    for (int n = 0; n < 4; ++n)
      #pragma unroll
      for (int r = 0; r < 8; ++r) {
        int row = wave*16 + r + (hiH ? 8 : 0);
        float v = acc[n][r];
        sH[row*HSTR + n*16 + lhalf] = (_Float16)(v > 0.f ? v : 0.f);
      }
  }
  __syncthreads();

  // ======== Layer 2: C2[128,128] = H[128,64] @ W2 + b2 (aliases sX) ========
  {
    v8f acc[8];
    #pragma unroll
    for (int n = 0; n < 8; ++n) {
      float bv = sF[O_AEB2 + n*16 + lhalf];
      v8f t = {bv,bv,bv,bv,bv,bv,bv,bv};
      acc[n] = t;
    }
    #pragma unroll
    for (int k0 = 0; k0 < AE_HID; k0 += 32) {
      const _Float16* pa = sH + (wave*16 + lhalf)*HSTR + k0 + (hiH ? 8 : 0);
      v16h a = cat16(*(const v8h*)pa, *(const v8h*)(pa + 16));
      #pragma unroll
      for (int n = 0; n < 8; ++n) {
        const _Float16* pb = sW2 + (n*16 + lhalf)*HSTR + k0 + (hiH ? 8 : 0);
        v16h b = cat16(*(const v8h*)pb, *(const v8h*)(pb + 16));
        acc[n] = __builtin_amdgcn_wmma_f32_16x16x32_f16(false, a, false, b,
                                                        (short)0, acc[n], false, false);
      }
    }
    _Float16* sC2 = sX;  // X dead after layer1 barrier
    #pragma unroll
    for (int n = 0; n < 8; ++n)
      #pragma unroll
      for (int r = 0; r < 8; ++r) {
        int row = wave*16 + r + (hiH ? 8 : 0);
        sC2[row*XSTR + n*16 + lhalf] = (_Float16)acc[n][r];
      }
  }
  __syncthreads();

  // ======== Trunk: one thread per query row ========
  if (tid < TILE_M) {
    long long gr = row0 + tid;
    if (gr < nrows) {
      const _Float16* sC2 = sX;
      float a1[H1D];   // widest temp

      // Linear 128->25 + BN + LeakyReLU
      float h0[HIDDEN];
      #pragma unroll
      for (int j = 0; j < HIDDEN; ++j) h0[j] = sF[O_BIN + j];
      for (int c = 0; c < INPUT_DIM; ++c) {
        float xv = (float)sC2[tid*XSTR + c];
        #pragma unroll
        for (int j = 0; j < HIDDEN; ++j) h0[j] += xv * sF[O_WIN + c*HIDDEN + j];
      }
      #pragma unroll
      for (int j = 0; j < HIDDEN; ++j) {
        float t = (h0[j] - sF[O_MIN + j]) * sF[O_GIN + j] *
                  __frsqrt_rn(sF[O_VIN + j] + BN_EPS) + sF[O_BEIN + j];
        h0[j] = (t >= 0.f) ? t : 0.01f * t;
      }
      // Linear 25->40 + BN + LeakyReLU
      #pragma unroll
      for (int j = 0; j < H1D; ++j) a1[j] = sF[O_BH1 + j];
      #pragma unroll
      for (int c = 0; c < HIDDEN; ++c) {
        float xv = h0[c];
        #pragma unroll
        for (int j = 0; j < H1D; ++j) a1[j] += xv * sF[O_WH1 + c*H1D + j];
      }
      #pragma unroll
      for (int j = 0; j < H1D; ++j) {
        float t = (a1[j] - sF[O_MH1 + j]) * sF[O_GH1 + j] *
                  __frsqrt_rn(sF[O_VH1 + j] + BN_EPS) + sF[O_BEH1 + j];
        a1[j] = (t >= 0.f) ? t : 0.01f * t;
      }
      // Linear 40->25 + BN + LeakyReLU
      float h2[H2D];
      #pragma unroll
      for (int j = 0; j < H2D; ++j) h2[j] = sF[O_BH2 + j];
      #pragma unroll
      for (int c = 0; c < H1D; ++c) {
        float xv = a1[c];
        #pragma unroll
        for (int j = 0; j < H2D; ++j) h2[j] += xv * sF[O_WH2 + c*H2D + j];
      }
      #pragma unroll
      for (int j = 0; j < H2D; ++j) {
        float t = (h2[j] - sF[O_MH2 + j]) * sF[O_GH2 + j] *
                  __frsqrt_rn(sF[O_VH2 + j] + BN_EPS) + sF[O_BEH2 + j];
        h2[j] = (t >= 0.f) ? t : 0.01f * t;
      }
      // Linear 25->10
      float e[OUTD];
      #pragma unroll
      for (int o = 0; o < OUTD; ++o) e[o] = sF[O_BOUT + o];
      #pragma unroll
      for (int c = 0; c < H2D; ++c) {
        float xv = h2[c];
        #pragma unroll
        for (int o = 0; o < OUTD; ++o) e[o] += xv * sF[O_WOUT + c*OUTD + o];
      }

      if (mode == 0) {
        #pragma unroll
        for (int o = 0; o < OUTD; ++o) emb_out[gr*OUTD + o] = e[o];
      } else {
        float nn = 0.f;
        #pragma unroll
        for (int o = 0; o < OUTD; ++o) nn += e[o]*e[o];
        float inv = 1.f / fmaxf(sqrtf(nn), COS_EPS);
        #pragma unroll
        for (int c = 0; c < NCLASS; ++c) {
          float d = 0.f;
          #pragma unroll
          for (int o = 0; o < OUTD; ++o) d += e[o] * sF[O_PN + c*OUTD + o];
          cos_out[gr*NCLASS + c] = d * inv;
        }
      }
    }
  }
}

// class prototypes (mean over contiguous blocks) + L2-normalize
__global__ void proto_kernel(const float* __restrict__ s_emb, int srows,
                             float* __restrict__ pn)
{
  __shared__ float proto[NCLASS][OUTD];
  int t = threadIdx.x;
  int every = srows / NCLASS;
  if (t < NCLASS*OUTD) {
    int c = t / OUTD, o = t % OUTD;
    float s = 0.f;
    for (int i = 0; i < every; ++i) s += s_emb[(c*every + i)*OUTD + o];
    proto[c][o] = s / (float)every;
  }
  __syncthreads();
  if (t < NCLASS) {
    float nn = 0.f;
    #pragma unroll
    for (int o = 0; o < OUTD; ++o) nn += proto[t][o]*proto[t][o];
    float inv = 1.f / fmaxf(sqrtf(nn), COS_EPS);
    #pragma unroll
    for (int o = 0; o < OUTD; ++o) pn[t*OUTD + o] = proto[t][o] * inv;
  }
}

extern "C" void kernel_launch(void* const* d_in, const int* in_sizes, int n_in,
                              void* d_out, int out_size, void* d_ws, size_t ws_size,
                              hipStream_t stream) {
  (void)n_in; (void)out_size; (void)ws_size;
  const float* support = (const float*)d_in[0];
  const float* query   = (const float*)d_in[1];
  // d_in[2] = query_label (unused by reference output)
  Params p;
  p.ae_w1 = (const float*)d_in[3];  p.ae_b1 = (const float*)d_in[4];
  p.ae_w2 = (const float*)d_in[5];  p.ae_b2 = (const float*)d_in[6];
  p.w_in  = (const float*)d_in[7];  p.b_in  = (const float*)d_in[8];
  p.g_in  = (const float*)d_in[9];  p.be_in = (const float*)d_in[10];
  p.m_in  = (const float*)d_in[11]; p.v_in  = (const float*)d_in[12];
  p.w_h1  = (const float*)d_in[13]; p.b_h1  = (const float*)d_in[14];
  p.g_h1  = (const float*)d_in[15]; p.be_h1 = (const float*)d_in[16];
  p.m_h1  = (const float*)d_in[17]; p.v_h1  = (const float*)d_in[18];
  p.w_h2  = (const float*)d_in[19]; p.b_h2  = (const float*)d_in[20];
  p.g_h2  = (const float*)d_in[21]; p.be_h2 = (const float*)d_in[22];
  p.m_h2  = (const float*)d_in[23]; p.v_h2  = (const float*)d_in[24];
  p.w_out = (const float*)d_in[25]; p.b_out = (const float*)d_in[26];

  int srows = in_sizes[0] / INPUT_DIM;   // 4096
  int qrows = in_sizes[1] / INPUT_DIM;   // 1,000,000

  float* s_emb = (float*)d_ws;                 // [srows][10]
  float* pn    = s_emb + (size_t)srows*OUTD;   // [2][10]
  float* out   = (float*)d_out;                // [qrows][2]

  int sgrid = (srows + TILE_M - 1) / TILE_M;
  int qgrid = (qrows + TILE_M - 1) / TILE_M;

  embed_kernel<<<sgrid, 256, LDS_BYTES, stream>>>(support, srows, p, s_emb, nullptr, nullptr, 0);
  proto_kernel<<<1, 64, 0, stream>>>(s_emb, srows, pn);
  embed_kernel<<<qgrid, 256, LDS_BYTES, stream>>>(query, qrows, p, nullptr, pn, out, 1);
}
